// SplitModule_59966333387115
// MI455X (gfx1250) — matrix operand order repaired and checked by reading, not compiled, not run on hardware
//
#include <hip/hip_runtime.h>
#include <hip/hip_bf16.h>
#include <stdint.h>

// ---------------------------------------------------------------------------
// MoE split-linear for MI455X (gfx1250, wave32, WMMA).
// Grouped-GEMM: sort tokens by expert, dense 16-row tiles, bf16 hi/lo split
// (3x v_wmma_f32_16x16x32_bf16) for ~f32 accuracy at bf16 matrix throughput.
// Round 4: bias folded into the store epilogue so the WMMA chain never waits
// on the per-column-pair bias loads (acc zero-init is immediate-only).
// ---------------------------------------------------------------------------

typedef __attribute__((ext_vector_type(16))) __bf16 v16bf;
typedef __attribute__((ext_vector_type(8)))  float  v8f;

#define NEXP 8
#define DIN  128
#define DOUT 128
#define TPB_TILES 64   // token-tiles (16 rows each) handled per block
#define WAVES 8        // 256 threads = 8 wave32

// ---- routing phase ---------------------------------------------------------

__global__ void k_zero(uint32_t* counts, uint32_t* running) {
    int i = threadIdx.x;
    if (i < NEXP) { counts[i] = 0u; running[i] = 0u; }
}

__global__ void k_count(const int* __restrict__ inds, int n, uint32_t* counts) {
    int i = blockIdx.x * blockDim.x + threadIdx.x;
    if (i < n) atomicAdd(&counts[inds[i] & 7], 1u);
}

// single thread: 16-aligned exclusive scan of 8 counts + sentinel-fill padding
__global__ void k_scan(const uint32_t* __restrict__ counts, uint32_t* seg,
                       uint32_t* perm) {
    if (threadIdx.x == 0) {
        uint32_t s = 0;
        for (int e = 0; e < NEXP; e++) {
            seg[e] = s;
            uint32_t c  = counts[e];
            uint32_t cr = (c + 15u) & ~15u;
            for (uint32_t j = c; j < cr; j++) perm[s + j] = 0xFFFFFFFFu;
            s += cr;
        }
        seg[NEXP] = s;
    }
}

__global__ void k_scatter(const int* __restrict__ inds, int n,
                          const uint32_t* __restrict__ seg,
                          uint32_t* running, uint32_t* perm) {
    int i = blockIdx.x * blockDim.x + threadIdx.x;
    if (i < n) {
        int e = inds[i] & 7;
        uint32_t slot = atomicAdd(&running[e], 1u);
        perm[seg[e] + slot] = (uint32_t)i;
    }
}

// W[e][k][n] (f32) -> transposed bf16 hi/lo: whiT/wloT[e][n][k]
__global__ void k_convw(const float* __restrict__ W,
                        __bf16* __restrict__ whiT, __bf16* __restrict__ wloT) {
    int idx = blockIdx.x * blockDim.x + threadIdx.x;
    if (idx >= NEXP * DIN * DOUT) return;
    int e  = idx >> 14;
    int r  = idx & 16383;
    int k  = r >> 7;
    int nn = r & 127;
    float w  = W[(size_t)e * DIN * DOUT + (size_t)k * DOUT + nn];
    __bf16 h = (__bf16)w;
    float lo = w - (float)h;
    size_t o = (size_t)e * DIN * DOUT + (size_t)nn * DIN + k;
    whiT[o] = h;
    wloT[o] = (__bf16)lo;
}

// ---- grouped GEMM phase ----------------------------------------------------
// grid = (ceil(maxTiles/TPB_TILES), NEXP). Each block: stage expert's B (hi+lo,
// 64 KB bf16, [n][k]) into LDS, then each wave computes 16x128 output tiles.

__global__ __launch_bounds__(256)
void k_moe_gemm(const float* __restrict__ feat, const float* __restrict__ bias,
                const uint32_t* __restrict__ seg, const uint32_t* __restrict__ perm,
                const __bf16* __restrict__ whiT, const __bf16* __restrict__ wloT,
                float* __restrict__ out) {
    __shared__ __bf16 sBhi[DOUT * DIN];   // 32 KB, [n][k]
    __shared__ __bf16 sBlo[DOUT * DIN];   // 32 KB

    int e = blockIdx.y;
    uint32_t s0 = seg[e], s1 = seg[e + 1];
    uint32_t ntiles = (s1 - s0) >> 4;
    uint32_t tbase  = (uint32_t)blockIdx.x * TPB_TILES;
    if (tbase >= ntiles) return;              // uniform per block

    // stage B hi/lo into LDS (L2 -> LDS, amortized over up to 64 tiles)
    {
        const float4* shi = (const float4*)(whiT + (size_t)e * DIN * DOUT);
        const float4* slo = (const float4*)(wloT + (size_t)e * DIN * DOUT);
        float4* dhi = (float4*)sBhi;
        float4* dlo = (float4*)sBlo;
        for (int i = threadIdx.x; i < (DOUT * DIN * 2) / 16; i += 256) {
            dhi[i] = shi[i];
            dlo[i] = slo[i];
        }
    }
    __syncthreads();

    int wave   = threadIdx.x >> 5;
    int lane   = threadIdx.x & 31;
    int laneHi = (lane & 16) ? 1 : 0;     // upper half-wave
    int l15    = lane & 15;

    const float* biasE = bias + e * DOUT + l15;

    uint32_t tEnd = tbase + TPB_TILES;
    if (tEnd > ntiles) tEnd = ntiles;

#pragma unroll 1
    for (uint32_t t = tbase + wave; t < tEnd; t += WAVES) {
        uint32_t pbase = s0 + (t << 4);
        // lane l (and l+16) holds perm entry for tile row l15
        uint32_t pv   = perm[pbase + l15];
        uint32_t rowA = (pv == 0xFFFFFFFFu) ? 0u : pv;   // padding reads row 0
        const float* arow = feat + (size_t)rowA * DIN;

        // output row index per C/D VGPR slot j (M = j + 8*laneHi)
        uint32_t prow[8];
#pragma unroll
        for (int j = 0; j < 8; j++) prow[j] = __shfl(pv, j + (laneHi ? 8 : 0), 32);

        // ---- load + split A for all 4 k-chunks (held in registers) --------
        // A k order used here, contiguous, same as B (16x32, M=lane&15):
        //  lanes 0-15: K = c*32 + 0..15
        //  lanes16-31: K = c*32 + 16..31
        v16bf ahi[4], alo[4];
#pragma unroll
        for (int c = 0; c < 4; c++) {
            int kb = c * 32 + (laneHi ? 16 : 0);
            float4 a0 = *(const float4*)(arow + kb);
            float4 a1 = *(const float4*)(arow + kb + 4);
            float4 a2 = *(const float4*)(arow + kb + 8);
            float4 a3 = *(const float4*)(arow + kb + 12);
            float af[16] = {a0.x, a0.y, a0.z, a0.w, a1.x, a1.y, a1.z, a1.w,
                            a2.x, a2.y, a2.z, a2.w, a3.x, a3.y, a3.z, a3.w};
#pragma unroll
            for (int i = 0; i < 16; i++) {
                float f  = af[i];
                __bf16 h = (__bf16)f;
                ahi[c][i] = h;
                alo[c][i] = (__bf16)(f - (float)h);
            }
        }

        // ---- column blocks in pairs, loop kept ROLLED to bound register
        // pressure (only acc0/acc1 + 4 transient B vectors live) ------------
#pragma unroll 1
        for (int nbp = 0; nbp < 4; nbp++) {
            const int nb0 = nbp * 2, nb1 = nbp * 2 + 1;
            // bias loads issue here, consumed only in the epilogue ~24 WMMAs
            // later -> latency fully hidden behind the matrix chain
            float bv0 = biasE[nb0 * 16];
            float bv1 = biasE[nb1 * 16];
            v8f acc0 = {}, acc1 = {};   // immediate zero-init, no memory wait

#pragma unroll
            for (int c = 0; c < 4; c++) {
                // B k order used here, contiguous, same as A (32x16, N=lane&15):
                //  lanes 0-15: K = c*32 + 0..15 ; lanes 16-31: + 16..31
                int kB = c * 32 + (laneHi ? 16 : 0);
                int n0 = nb0 * 16 + l15;
                int n1 = nb1 * 16 + l15;
                v16bf bhi0 = *(const v16bf*)&sBhi[n0 * DIN + kB];
                v16bf bhi1 = *(const v16bf*)&sBhi[n1 * DIN + kB];
                v16bf blo0 = *(const v16bf*)&sBlo[n0 * DIN + kB];
                v16bf blo1 = *(const v16bf*)&sBlo[n1 * DIN + kB];
                // f32-accurate split: Ahi*Bhi + Ahi*Blo + Alo*Bhi
                acc0 = __builtin_amdgcn_wmma_f32_16x16x32_bf16(
                    false, ahi[c], false, bhi0, (short)0, acc0, false, false);
                acc1 = __builtin_amdgcn_wmma_f32_16x16x32_bf16(
                    false, ahi[c], false, bhi1, (short)0, acc1, false, false);
                acc0 = __builtin_amdgcn_wmma_f32_16x16x32_bf16(
                    false, ahi[c], false, blo0, (short)0, acc0, false, false);
                acc1 = __builtin_amdgcn_wmma_f32_16x16x32_bf16(
                    false, ahi[c], false, blo1, (short)0, acc1, false, false);
                acc0 = __builtin_amdgcn_wmma_f32_16x16x32_bf16(
                    false, alo[c], false, bhi0, (short)0, acc0, false, false);
                acc1 = __builtin_amdgcn_wmma_f32_16x16x32_bf16(
                    false, alo[c], false, bhi1, (short)0, acc1, false, false);
            }

            // scatter + bias add (lanes 0-15 cover 64B of each row)
#pragma unroll
            for (int j = 0; j < 8; j++) {
                if (prow[j] != 0xFFFFFFFFu) {
                    float* orow = out + (size_t)prow[j] * DOUT + l15;
                    orow[nb0 * 16] = acc0[j] + bv0;
                    orow[nb1 * 16] = acc1[j] + bv1;
                }
            }
        }
    }
}

// ---------------------------------------------------------------------------

extern "C" void kernel_launch(void* const* d_in, const int* in_sizes, int n_in,
                              void* d_out, int out_size, void* d_ws, size_t ws_size,
                              hipStream_t stream) {
    const float* feat = (const float*)d_in[0];
    const int*   inds = (const int*)d_in[1];
    const float* W    = (const float*)d_in[2];
    const float* bias = (const float*)d_in[3];
    float* out = (float*)d_out;
    int n = in_sizes[1];   // one routing index per token

    // workspace layout
    char* ws = (char*)d_ws;
    uint32_t* counts  = (uint32_t*)(ws + 0);
    uint32_t* running = (uint32_t*)(ws + 64);
    uint32_t* seg     = (uint32_t*)(ws + 128);
    uint32_t* perm    = (uint32_t*)(ws + 256);
    size_t permBytes  = ((size_t)n + 8 * 16) * sizeof(uint32_t);
    size_t off = 256 + ((permBytes + 255) & ~(size_t)255);
    __bf16* whiT = (__bf16*)(ws + off);
    __bf16* wloT = (__bf16*)(ws + off + (size_t)NEXP * DIN * DOUT * 2);

    k_zero<<<1, 64, 0, stream>>>(counts, running);
    k_count<<<(n + 255) / 256, 256, 0, stream>>>(inds, n, counts);
    k_scan<<<1, 1, 0, stream>>>(counts, seg, perm);
    k_scatter<<<(n + 255) / 256, 256, 0, stream>>>(inds, n, seg, running, perm);
    k_convw<<<(NEXP * DIN * DOUT + 255) / 256, 256, 0, stream>>>(W, whiT, wloT);

    int maxTiles = (n + 15) / 16;   // worst case: one expert owns everything
    dim3 grid((maxTiles + TPB_TILES - 1) / TPB_TILES, NEXP, 1);
    k_moe_gemm<<<grid, 256, 0, stream>>>(feat, bias, seg, perm, whiT, wloT, out);
}
